// MultiHeadAttention_43946105373152
// MI455X (gfx1250) — matrix operand order repaired and checked by reading, not compiled, not run on hardware
//
#include <hip/hip_runtime.h>
#include <hip/hip_bf16.h>
#include <math.h>

#define DM 512
#define NH 8
#define DK 64
#define NB 4
#define SS 2048
#define MROWS (NB * SS)   // 8192

typedef __bf16 bf16_t;
typedef __attribute__((ext_vector_type(16))) __bf16 v16bf;
typedef __attribute__((ext_vector_type(8)))  __bf16 v8bf;
typedef __attribute__((ext_vector_type(8)))  float  v8f;

__device__ __forceinline__ bf16_t to_bf16(float f) {
  union { float f; unsigned int u; } x; x.f = f;
  unsigned int r = x.u + 0x7FFFu + ((x.u >> 16) & 1u);
  union { unsigned short u; bf16_t b; } o;
  o.u = (unsigned short)(r >> 16);
  return o.b;
}

// A-operand fragment: 16x32 bf16 (MxK), row-major source with leading dim lda.
// ISA layout: lanes 0-15 hold row M=lane, K=0..7 & 16..23; lanes 16-31 hold
// row M=lane-16, K=8..15 & 24..31.
__device__ __forceinline__ v16bf load_a_frag(const bf16_t* A, int lda) {
  const int lane = threadIdx.x & 31;
  const int half = lane >> 4;
  const int r    = lane & 15;
  union { v16bf v; v8bf h[2]; } f;
  const bf16_t* p = A + (size_t)r * lda + half * 8;
  f.h[0] = *(const v8bf*)(p);
  f.h[1] = *(const v8bf*)(p + 16);
  return f.v;
}

// B-operand fragment: 32x16 bf16 (KxN). Element (k,n) read from Bp[n*ldb + k]
// (column n contiguous along K). ISA layout: lane = N; lanes 0-15 hold
// K=0..7 & 16..23, lanes 16-31 hold K=8..15 & 24..31.
__device__ __forceinline__ v16bf load_b_frag(const bf16_t* Bp, int ldb) {
  const int lane = threadIdx.x & 31;
  const int half = lane >> 4;
  const int n    = lane & 15;
  union { v16bf v; v8bf h[2]; } f;
  const bf16_t* p = Bp + (size_t)n * ldb + half * 8;
  f.h[0] = *(const v8bf*)(p);
  f.h[1] = *(const v8bf*)(p + 16);
  return f.v;
}

__device__ __forceinline__ v8f wmma_bf16(v16bf a, v16bf b, v8f c) {
  return __builtin_amdgcn_wmma_f32_16x16x32_bf16(
      /*neg_a=*/false, a, /*neg_b=*/false, b,
      /*c_mod=*/(short)0, c, /*reuse_a=*/false, /*reuse_b=*/false);
}

// -------------------- fp32 -> bf16 conversion --------------------
__global__ void cvt_kernel(const float* __restrict__ src,
                           bf16_t* __restrict__ dst, int n) {
  int i = blockIdx.x * blockDim.x + threadIdx.x;
  if (i < n) dst[i] = to_bf16(src[i]);
}

// -------------------- GEMM: Y = X * W^T + bias --------------------
// X: [MROWS, DM] bf16 row-major. W: [DM, DM] bf16 row-major (nn.Linear weight).
// Each wave computes a 16x64 tile: A-frag loaded once per K-step and reused
// across 4 WMMAs (4 fp32 accumulators). Block = 4 waves -> 16x256 tile.
// mode 0: Yb bf16 row-major [MROWS, DM]
// mode 1: Yb bf16 head-transposed [B, H, DK, S]  (used for V)
// mode 2: Yf fp32 row-major [MROWS, DM]          (final output)
__global__ void gemm_bias_kernel(const bf16_t* __restrict__ X,
                                 const bf16_t* __restrict__ W,
                                 const float* __restrict__ bias,
                                 bf16_t* __restrict__ Yb,
                                 float* __restrict__ Yf,
                                 int mode) {
  const int wave = threadIdx.x >> 5;
  const int lane = threadIdx.x & 31;
  const int half = lane >> 4;
  const int nidx = lane & 15;
  const int m0 = blockIdx.x * 16;
  const int n0 = blockIdx.y * 256 + wave * 64;

  v8f acc[4];
  #pragma unroll
  for (int t = 0; t < 4; ++t)
    #pragma unroll
    for (int i = 0; i < 8; ++i) acc[t][i] = 0.0f;

  for (int kk = 0; kk < DM; kk += 32) {
    const v16bf a = load_a_frag(X + (size_t)m0 * DM + kk, DM);
    #pragma unroll
    for (int t = 0; t < 4; ++t) {
      v16bf b = load_b_frag(W + (size_t)(n0 + t * 16) * DM + kk, DM);
      acc[t] = wmma_bf16(a, b, acc[t]);
    }
  }

  #pragma unroll
  for (int t = 0; t < 4; ++t) {
    const int col = n0 + t * 16 + nidx;
    const float bval = bias[col];
    #pragma unroll
    for (int vr = 0; vr < 8; ++vr) {
      const int row = m0 + half * 8 + vr;
      const float y = acc[t][vr] + bval;
      if (mode == 0) {
        Yb[(size_t)row * DM + col] = to_bf16(y);
      } else if (mode == 1) {
        const int bbi = row / SS, si = row % SS;
        const int hh = col / DK, dd = col % DK;
        Yb[(((size_t)(bbi * NH + hh) * DK + dd) * SS) + si] = to_bf16(y);
      } else {
        Yf[(size_t)row * DM + col] = y;
      }
    }
  }
}

// -------------------- Flash attention per (b, h, 16-query tile) ----------
// Qp, Kp: [B*S, DM] bf16 (head h at columns h*DK .. h*DK+63)
// Vt    : [B, H, DK, S] bf16 (seq contiguous)
// Ctx   : [B*S, DM] bf16 (concat of heads)
__global__ void attn_kernel(const bf16_t* __restrict__ Qp,
                            const bf16_t* __restrict__ Kp,
                            const bf16_t* __restrict__ Vt,
                            const int* __restrict__ mask,
                            bf16_t* __restrict__ Ctx) {
  const int wave = threadIdx.x >> 5;
  const int lane = threadIdx.x & 31;
  const int half = lane >> 4;
  const int nidx = lane & 15;

  const int tiles_per_bh = SS / 64;               // 32 blocks of 64 queries
  const int bh = blockIdx.x / tiles_per_bh;
  const int b  = bh / NH;
  const int h  = bh % NH;
  const int q0 = (blockIdx.x % tiles_per_bh) * 64 + wave * 16;

  const bf16_t* Qbase = Qp + (size_t)(b * SS + q0) * DM + h * DK;
  const bf16_t* Kbase = Kp + (size_t)(b * SS) * DM + h * DK;
  const bf16_t* Vbase = Vt + (size_t)bh * DK * SS;

  // per-wave private staging buffer; LDS ops are in-order within a wave
  // (DScnt ordering) so no block barrier is needed -- compiler-level fences
  // only, to keep the store->load program order.
  __shared__ __align__(32) bf16_t ldsP[4][16 * 32];
  bf16_t* myP = ldsP[wave];

  const v16bf qf0 = load_a_frag(Qbase, DM);
  const v16bf qf1 = load_a_frag(Qbase + 32, DM);

  v8f acc[4];
  float m[8], l[8];
  #pragma unroll
  for (int t = 0; t < 4; ++t)
    #pragma unroll
    for (int i = 0; i < 8; ++i) acc[t][i] = 0.0f;
  #pragma unroll
  for (int vr = 0; vr < 8; ++vr) { m[vr] = -1e30f; l[vr] = 0.0f; }

  const float scale = 0.125f;  // 1/sqrt(64)

  for (int kv = 0; kv < SS; kv += 32) {
    // ---- scores: two 16x16 tiles, d_k=64 via two K=32 steps each ----
    v8f s0, s1;
    {
      v16bf kf0 = load_b_frag(Kbase + (size_t)kv * DM, DM);
      v16bf kf1 = load_b_frag(Kbase + (size_t)kv * DM + 32, DM);
      v8f z;
      #pragma unroll
      for (int i = 0; i < 8; ++i) z[i] = 0.0f;
      z = wmma_bf16(qf0, kf0, z);
      z = wmma_bf16(qf1, kf1, z);
      s0 = z;
    }
    {
      v16bf kf0 = load_b_frag(Kbase + (size_t)(kv + 16) * DM, DM);
      v16bf kf1 = load_b_frag(Kbase + (size_t)(kv + 16) * DM + 32, DM);
      v8f z;
      #pragma unroll
      for (int i = 0; i < 8; ++i) z[i] = 0.0f;
      z = wmma_bf16(qf0, kf0, z);
      z = wmma_bf16(qf1, kf1, z);
      s1 = z;
    }

    // ---- scale + mask ----
    const int kc0 = kv + nidx;
    const int kc1 = kv + 16 + nidx;
    #pragma unroll
    for (int vr = 0; vr < 8; ++vr) {
      const int qrow = q0 + half * 8 + vr;
      const size_t mrow = (size_t)qrow * SS;
      s0[vr] = mask[mrow + kc0] ? s0[vr] * scale : -1e9f;
      s1[vr] = mask[mrow + kc1] ? s1[vr] * scale : -1e9f;
    }

    // ---- online softmax (row stats live in 8 regs/lane; rows match C layout) ----
    #pragma unroll
    for (int vr = 0; vr < 8; ++vr) {
      float v = fmaxf(s0[vr], s1[vr]);
      v = fmaxf(v, __shfl_xor(v, 1, 32));
      v = fmaxf(v, __shfl_xor(v, 2, 32));
      v = fmaxf(v, __shfl_xor(v, 4, 32));
      v = fmaxf(v, __shfl_xor(v, 8, 32));
      const float mn = fmaxf(m[vr], v);
      const float corr = __expf(m[vr] - mn);
      m[vr] = mn;
      s0[vr] = __expf(s0[vr] - mn);
      s1[vr] = __expf(s1[vr] - mn);
      float rs = s0[vr] + s1[vr];
      rs += __shfl_xor(rs, 1, 32);
      rs += __shfl_xor(rs, 2, 32);
      rs += __shfl_xor(rs, 4, 32);
      rs += __shfl_xor(rs, 8, 32);
      l[vr] = l[vr] * corr + rs;
      acc[0][vr] *= corr; acc[1][vr] *= corr;
      acc[2][vr] *= corr; acc[3][vr] *= corr;
    }

    // ---- C-layout -> A-layout for P via LDS (16x32 bf16, 1KB/wave) ----
    #pragma unroll
    for (int vr = 0; vr < 8; ++vr) {
      const int prow = half * 8 + vr;
      myP[prow * 32 + nidx]      = to_bf16(s0[vr]);
      myP[prow * 32 + 16 + nidx] = to_bf16(s1[vr]);
    }
    __builtin_amdgcn_fence(__ATOMIC_RELEASE, "wavefront");
    const v16bf pf = load_a_frag(myP, 32);
    __builtin_amdgcn_fence(__ATOMIC_ACQUIRE, "wavefront");

    // ---- acc += P(16x32) @ V(32x64) : 4 WMMAs, V columns contiguous in seq ----
    #pragma unroll
    for (int t = 0; t < 4; ++t) {
      v16bf vf = load_b_frag(Vbase + (size_t)(t * 16) * SS + kv, SS);
      acc[t] = wmma_bf16(pf, vf, acc[t]);
    }
  }

  // ---- epilogue: normalize and store into concat buffer ----
  #pragma unroll
  for (int t = 0; t < 4; ++t) {
    #pragma unroll
    for (int vr = 0; vr < 8; ++vr) {
      const int row = q0 + half * 8 + vr;
      const int col = h * DK + t * 16 + nidx;
      Ctx[(size_t)(b * SS + row) * DM + col] = to_bf16(acc[t][vr] / l[vr]);
    }
  }
}

// -------------------- host launch --------------------
extern "C" void kernel_launch(void* const* d_in, const int* in_sizes, int n_in,
                              void* d_out, int out_size, void* d_ws, size_t ws_size,
                              hipStream_t stream) {
  const float* q    = (const float*)d_in[0];
  const float* k    = (const float*)d_in[1];
  const float* v    = (const float*)d_in[2];
  const int*   mask = (const int*)  d_in[3];
  const float* Wq   = (const float*)d_in[4];
  const float* bq   = (const float*)d_in[5];
  const float* Wk   = (const float*)d_in[6];
  const float* bk   = (const float*)d_in[7];
  const float* Wv   = (const float*)d_in[8];
  const float* bv   = (const float*)d_in[9];
  const float* Wo   = (const float*)d_in[10];
  const float* bo   = (const float*)d_in[11];
  float* out = (float*)d_out;

  char* ws = (char*)d_ws;
  size_t off = 0;
  auto carve = [&](size_t bytes) -> char* {
    char* p = ws + off;
    off += (bytes + 255) & ~(size_t)255;
    return p;
  };
  const size_t actBytes = (size_t)MROWS * DM * sizeof(bf16_t);  // 8 MB each
  const size_t wBytes   = (size_t)DM * DM * sizeof(bf16_t);

  bf16_t* qbf  = (bf16_t*)carve(actBytes);
  bf16_t* kbf  = (bf16_t*)carve(actBytes);
  bf16_t* vbf  = (bf16_t*)carve(actBytes);
  bf16_t* wqbf = (bf16_t*)carve(wBytes);
  bf16_t* wkbf = (bf16_t*)carve(wBytes);
  bf16_t* wvbf = (bf16_t*)carve(wBytes);
  bf16_t* wobf = (bf16_t*)carve(wBytes);
  bf16_t* Qp   = (bf16_t*)carve(actBytes);
  bf16_t* Kp   = (bf16_t*)carve(actBytes);
  bf16_t* Vt   = (bf16_t*)carve(actBytes);   // [B,H,DK,S]
  bf16_t* Ctx  = (bf16_t*)carve(actBytes);

  const int actN = MROWS * DM;  // 4,194,304
  const int wN   = DM * DM;     // 262,144
  cvt_kernel<<<dim3((actN + 255) / 256), 256, 0, stream>>>(q, qbf, actN);
  cvt_kernel<<<dim3((actN + 255) / 256), 256, 0, stream>>>(k, kbf, actN);
  cvt_kernel<<<dim3((actN + 255) / 256), 256, 0, stream>>>(v, vbf, actN);
  cvt_kernel<<<dim3((wN + 255) / 256),   256, 0, stream>>>(Wq, wqbf, wN);
  cvt_kernel<<<dim3((wN + 255) / 256),   256, 0, stream>>>(Wk, wkbf, wN);
  cvt_kernel<<<dim3((wN + 255) / 256),   256, 0, stream>>>(Wv, wvbf, wN);
  cvt_kernel<<<dim3((wN + 255) / 256),   256, 0, stream>>>(Wo, wobf, wN);

  dim3 gg(MROWS / 16, DM / 256);  // (512, 2), 4 waves/block, 16x64 tile per wave
  gemm_bias_kernel<<<gg, 128, 0, stream>>>(qbf, wqbf, bq, Qp, nullptr, 0);
  gemm_bias_kernel<<<gg, 128, 0, stream>>>(kbf, wkbf, bk, Kp, nullptr, 0);
  gemm_bias_kernel<<<gg, 128, 0, stream>>>(vbf, wvbf, bv, Vt, nullptr, 1);

  attn_kernel<<<dim3(NB * NH * (SS / 64)), 128, 0, stream>>>(Qp, Kp, Vt, mask, Ctx);

  gemm_bias_kernel<<<gg, 128, 0, stream>>>(Ctx, wobf, bo, nullptr, out, 2);
}
